// BiDAFBlock_42760694399499
// MI455X (gfx1250) — compile-verified, matrix operand order checked by reading, not run on hardware
//
#include <hip/hip_runtime.h>
#include <hip/hip_bf16.h>
#include <stdint.h>

// ---------------- problem constants ----------------
#define BB    32
#define CL    384
#define QL    64
#define HH    768
#define HH2   1536
#define HH4   3072
#define BC    (BB*CL)      // 12288
#define BQ    (BB*QL)      // 2048

typedef __bf16 bf16_t;
typedef __attribute__((ext_vector_type(16))) __bf16 v16bf;
typedef __attribute__((ext_vector_type(8)))  float  v8f;

__device__ __forceinline__ v8f vzero8() {
  v8f z;
#pragma unroll
  for (int i = 0; i < 8; ++i) z[i] = 0.f;
  return z;
}

__device__ __forceinline__ v8f wmma_bf16(v16bf a, v16bf b, v8f c) {
  // D = A(16x32 bf16) x B(32x16 bf16) + C(16x16 f32)
  return __builtin_amdgcn_wmma_f32_16x16x32_bf16(false, a, false, b, (short)0, c,
                                                 false, false);
}

__device__ __forceinline__ float fsigmoid(float x) { return 1.f / (1.f + __expf(-x)); }
__device__ __forceinline__ float ftanh(float x) {
  float e2 = __expf(2.f * x);
  return (e2 - 1.f) / (e2 + 1.f);
}

// ---------------- small prep kernels ----------------

__global__ void k_cvt_bf16(const float* __restrict__ in, bf16_t* __restrict__ out, size_t n) {
  size_t i = (size_t)blockIdx.x * blockDim.x + threadIdx.x;
  if (i < n) out[i] = (bf16_t)in[i];
}

// q prep: qw_bf[b,j,h] = bf16(q*w_cq[h]); qT_bf[b,h,j] = bf16(q)
__global__ void k_prep_q(const float* __restrict__ q, const float* __restrict__ wcq,
                         bf16_t* __restrict__ qw_bf, bf16_t* __restrict__ qT_bf) {
  size_t i = (size_t)blockIdx.x * blockDim.x + threadIdx.x;
  if (i >= (size_t)BQ * HH) return;
  int h = (int)(i % HH);
  size_t bj = i / HH;
  int j = (int)(bj % QL);
  int b = (int)(bj / QL);
  float v = q[i];
  qw_bf[i] = (bf16_t)(v * wcq[h]);
  qT_bf[((size_t)b * HH + h) * QL + j] = (bf16_t)v;
}

// out[r] = dot(X[r,:K], w) + bias[0]; one wave per row
__global__ void k_rowdot(const float* __restrict__ X, const float* __restrict__ w,
                         const float* __restrict__ bias, float* __restrict__ out,
                         int nrows, int K) {
  int r = (int)((blockIdx.x * blockDim.x + threadIdx.x) >> 5);
  int lane = threadIdx.x & 31;
  if (r >= nrows) return;
  const float* xr = X + (size_t)r * K;
  float s = 0.f;
  for (int k = lane; k < K; k += 32) s += xr[k] * w[k];
#pragma unroll
  for (int m = 16; m >= 1; m >>= 1) s += __shfl_xor(s, m, 32);
  if (lane == 0) out[r] = s + bias[0];
}

// ---------------- attention ----------------

__global__ void k_sim_wmma(const bf16_t* __restrict__ cbf, const bf16_t* __restrict__ qwbf,
                           const float* __restrict__ cvec, const float* __restrict__ qvec,
                           const float* __restrict__ bcq, float* __restrict__ s) {
  int b = blockIdx.x, mt = blockIdx.y, nt = blockIdx.z;
  int lane = threadIdx.x & 31;
  int ar = lane & 15, ak = (lane >> 4) * 16;
  const bf16_t* Ap = cbf  + ((size_t)b * CL + mt * 16 + ar) * HH + ak;
  const bf16_t* Bp = qwbf + ((size_t)b * QL + nt * 16 + ar) * HH + ak;
  v8f acc = vzero8();
#pragma unroll 4
  for (int k = 0; k < HH; k += 32) {
    v16bf a  = *(const v16bf*)(Ap + k);
    v16bf bm = *(const v16bf*)(Bp + k);
    acc = wmma_bf16(a, bm, acc);
  }
  int col = lane & 15, rb = (lane >> 4) * 8;
  float addq = qvec[b * QL + nt * 16 + col] + bcq[0];
#pragma unroll
  for (int e = 0; e < 8; ++e) {
    int row = mt * 16 + rb + e;
    s[((size_t)b * CL + row) * QL + nt * 16 + col] = acc[e] + cvec[b * CL + row] + addq;
  }
}

__global__ void k_softmax_rows(const float* __restrict__ s, bf16_t* __restrict__ abf,
                               float* __restrict__ smax) {
  int r = (int)((blockIdx.x * blockDim.x + threadIdx.x) >> 5);
  int lane = threadIdx.x & 31;
  if (r >= BC) return;
  const float* row = s + (size_t)r * QL;
  float v0 = row[lane], v1 = row[lane + 32];
  float mx = fmaxf(v0, v1);
#pragma unroll
  for (int m = 16; m >= 1; m >>= 1) mx = fmaxf(mx, __shfl_xor(mx, m, 32));
  float e0 = __expf(v0 - mx), e1 = __expf(v1 - mx);
  float sm = e0 + e1;
#pragma unroll
  for (int m = 16; m >= 1; m >>= 1) sm += __shfl_xor(sm, m, 32);
  float inv = 1.f / sm;
  abf[(size_t)r * QL + lane]      = (bf16_t)(e0 * inv);
  abf[(size_t)r * QL + lane + 32] = (bf16_t)(e1 * inv);
  if (lane == 0) smax[r] = mx;
}

__global__ void k_bw_softmax(const float* __restrict__ smax, float* __restrict__ bw) {
  int b = blockIdx.x, lane = threadIdx.x;
  float vals[CL / 32];
  float mx = -1e30f;
#pragma unroll
  for (int i = 0; i < CL / 32; ++i) {
    vals[i] = smax[b * CL + lane + i * 32];
    mx = fmaxf(mx, vals[i]);
  }
#pragma unroll
  for (int m = 16; m >= 1; m >>= 1) mx = fmaxf(mx, __shfl_xor(mx, m, 32));
  float sm = 0.f;
#pragma unroll
  for (int i = 0; i < CL / 32; ++i) { vals[i] = __expf(vals[i] - mx); sm += vals[i]; }
#pragma unroll
  for (int m = 16; m >= 1; m >>= 1) sm += __shfl_xor(sm, m, 32);
  float inv = 1.f / sm;
#pragma unroll
  for (int i = 0; i < CL / 32; ++i) bw[b * CL + lane + i * 32] = vals[i] * inv;
}

__global__ void k_q2c(const float* __restrict__ bw, const float* __restrict__ c,
                      float* __restrict__ q2c) {
  int b = blockIdx.x;
  int h = blockIdx.y * blockDim.x + threadIdx.x;
  float s = 0.f;
  for (int i = 0; i < CL; ++i) s += bw[b * CL + i] * c[((size_t)b * CL + i) * HH + h];
  q2c[b * HH + h] = s;
}

__global__ void k_c2q_wmma(const bf16_t* __restrict__ abf, const bf16_t* __restrict__ qTbf,
                           bf16_t* __restrict__ gbf) {
  int b = blockIdx.x, mt = blockIdx.y, nt = blockIdx.z;
  int lane = threadIdx.x & 31;
  int ar = lane & 15, ak = (lane >> 4) * 16;
  const bf16_t* Ap = abf  + ((size_t)b * CL + mt * 16 + ar) * QL + ak;
  const bf16_t* Bp = qTbf + ((size_t)b * HH + nt * 16 + ar) * QL + ak;
  v8f acc = vzero8();
#pragma unroll
  for (int k = 0; k < QL; k += 32) {
    v16bf a  = *(const v16bf*)(Ap + k);
    v16bf bm = *(const v16bf*)(Bp + k);
    acc = wmma_bf16(a, bm, acc);
  }
  int col = lane & 15, rb = (lane >> 4) * 8;
#pragma unroll
  for (int e = 0; e < 8; ++e) {
    int row = mt * 16 + rb + e;
    gbf[((size_t)b * CL + row) * HH4 + HH + nt * 16 + col] = (bf16_t)acc[e];
  }
}

__global__ void k_concat_g(const float* __restrict__ c, const float* __restrict__ q2c,
                           bf16_t* __restrict__ gbf) {
  size_t idx = (size_t)blockIdx.x * blockDim.x + threadIdx.x;
  if (idx >= (size_t)BC * HH) return;
  int h = (int)(idx % HH);
  size_t r = idx / HH;           // r = b*CL + i
  int b = (int)(r / CL);
  float cv = c[idx];
  bf16_t* grow = gbf + r * HH4;
  float t = (float)grow[HH + h];
  grow[h]           = (bf16_t)cv;
  grow[2 * HH + h]  = (bf16_t)(cv * t);
  grow[3 * HH + h]  = (bf16_t)(cv * q2c[b * HH + h]);
}

// ---------------- LSTM input-projection GEMM ----------------
// out[r,n] = sum_k A[r,k]*W[n,k] + bias[n].
// Block: 8 waves; macro-tile M=32 x N=512. Wave: 2 M-tiles x 4 N-tiles.
// A macro-tile chunk (32 rows x 32 k) staged into LDS with async-to-LDS copies,
// double-buffered against the WMMA stream (ASYNCcnt + barrier pipelining).
__global__ __launch_bounds__(256) void k_ih_gemm(const bf16_t* __restrict__ A,
                                                 const bf16_t* __restrict__ W,
                                                 const float* __restrict__ bias,
                                                 float* __restrict__ out, int Kdim) {
  __shared__ __align__(32) bf16_t Ash[2][32 * 32];   // 2 x 2KB
  int wv = threadIdx.x >> 5, lane = threadIdx.x & 31;
  int m0 = blockIdx.x * 32;
  int n0 = blockIdx.y * 512 + wv * 64;
  int ar = lane & 15, ak = (lane >> 4) * 16;
  int col = lane & 15, rb = (lane >> 4) * 8;

  const bf16_t* Bp[4];
#pragma unroll
  for (int j = 0; j < 4; ++j)
    Bp[j] = W + (size_t)(n0 + j * 16 + ar) * Kdim + ak;

  // async copy of one 32x32 A chunk into Ash[buf]: 128 lanes x 16B
  auto prefetchA = [&](int k, int buf) {
    if (threadIdx.x < 128) {
      int row = threadIdx.x >> 2;
      int ko  = (threadIdx.x & 3) * 8;
      uint64_t ga  = (uint64_t)(uintptr_t)(A + (size_t)(m0 + row) * Kdim + k + ko);
      uint32_t lds = (uint32_t)(uintptr_t)(&Ash[buf][row * 32 + ko]);
      asm volatile("global_load_async_to_lds_b128 %0, %1, off"
                   :: "v"(lds), "v"(ga) : "memory");
    }
  };

  v8f acc[2][4];
#pragma unroll
  for (int i = 0; i < 2; ++i)
#pragma unroll
    for (int j = 0; j < 4; ++j) acc[i][j] = vzero8();

  prefetchA(0, 0);
  asm volatile("s_wait_asynccnt 0x0" ::: "memory");
  __syncthreads();

  int nk = Kdim / 32;
  for (int kc = 0; kc < nk; ++kc) {
    int k = kc * 32;
    if (kc + 1 < nk) prefetchA(k + 32, (kc + 1) & 1);
    const bf16_t* abase = &Ash[kc & 1][0];
    v16bf a0 = *(const v16bf*)(abase + ar * 32 + ak);
    v16bf a1 = *(const v16bf*)(abase + (16 + ar) * 32 + ak);
#pragma unroll
    for (int j = 0; j < 4; ++j) {
      v16bf bm = *(const v16bf*)(Bp[j] + k);
      acc[0][j] = wmma_bf16(a0, bm, acc[0][j]);
      acc[1][j] = wmma_bf16(a1, bm, acc[1][j]);
    }
    asm volatile("s_wait_asynccnt 0x0" ::: "memory");
    __syncthreads();
  }

#pragma unroll
  for (int i = 0; i < 2; ++i)
#pragma unroll
    for (int j = 0; j < 4; ++j) {
      float bv = bias[n0 + j * 16 + col];
#pragma unroll
      for (int e = 0; e < 8; ++e)
        out[(size_t)(m0 + i * 16 + rb + e) * HH4 + n0 + j * 16 + col] = acc[i][j][e] + bv;
    }
}

// ---------------- LSTM recurrence: both directions concurrently ----------------
// grid = 2 workgroups (blockIdx.x = direction), 32 waves each.
// Wave owns 3 (m-tile, n-col) pairs; cell state in regs, h in LDS (48 KB).
__global__ __launch_bounds__(1024) void k_lstm_steps2(const float* __restrict__ xg_f,
                                                      const float* __restrict__ xg_b,
                                                      const bf16_t* __restrict__ whh_f,
                                                      const bf16_t* __restrict__ whh_b,
                                                      float* __restrict__ out) {
  __shared__ bf16_t hsh[32 * HH];
  int dir = blockIdx.x;
  const float*  xg  = dir ? xg_b  : xg_f;
  const bf16_t* whh = dir ? whh_b : whh_f;
  int wv = threadIdx.x >> 5, lane = threadIdx.x & 31;
  int ar = lane & 15, ak = (lane >> 4) * 16;
  int col = lane & 15, rb = (lane >> 4) * 8;
  for (int i = threadIdx.x; i < 32 * HH; i += 1024) hsh[i] = (bf16_t)0.f;
  float cst[3][8];
#pragma unroll
  for (int p = 0; p < 3; ++p)
#pragma unroll
    for (int e = 0; e < 8; ++e) cst[p][e] = 0.f;
  __syncthreads();
  int dirOff = dir ? HH : 0;

  for (int t = 0; t < CL; ++t) {
    int tt = dir ? (CL - 1 - t) : t;
    float ht[3][8];
#pragma unroll
    for (int p = 0; p < 3; ++p) {
      int pair = wv * 3 + p;
      int m0 = (pair & 1) * 16;
      int n0 = (pair >> 1) * 16;
      v8f acc[4];
#pragma unroll
      for (int gq = 0; gq < 4; ++gq) acc[gq] = vzero8();
      for (int k = 0; k < HH; k += 32) {
        v16bf a = *(const v16bf*)(&hsh[(m0 + ar) * HH + k + ak]);
#pragma unroll
        for (int gq = 0; gq < 4; ++gq) {
          v16bf bm = *(const v16bf*)(whh + ((size_t)(gq * HH + n0 + ar)) * HH + k + ak);
          acc[gq] = wmma_bf16(a, bm, acc[gq]);
        }
      }
#pragma unroll
      for (int e = 0; e < 8; ++e) {
        int bidx = m0 + rb + e;
        size_t xrow = ((size_t)bidx * CL + tt) * HH4;
        float gi = acc[0][e] + xg[xrow + 0 * HH + n0 + col];
        float gf = acc[1][e] + xg[xrow + 1 * HH + n0 + col];
        float gg = acc[2][e] + xg[xrow + 2 * HH + n0 + col];
        float go = acc[3][e] + xg[xrow + 3 * HH + n0 + col];
        float iv = fsigmoid(gi), fv = fsigmoid(gf), gv = ftanh(gg), ov = fsigmoid(go);
        float cn = fv * cst[p][e] + iv * gv;
        cst[p][e] = cn;
        ht[p][e] = ov * ftanh(cn);
      }
    }
    __syncthreads();   // all reads of hsh for this step done
#pragma unroll
    for (int p = 0; p < 3; ++p) {
      int pair = wv * 3 + p;
      int m0 = (pair & 1) * 16;
      int n0 = (pair >> 1) * 16;
#pragma unroll
      for (int e = 0; e < 8; ++e) {
        int bidx = m0 + rb + e;
        int h = n0 + col;
        hsh[bidx * HH + h] = (bf16_t)ht[p][e];
        out[((size_t)bidx * CL + tt) * HH2 + dirOff + h] = ht[p][e];
      }
    }
    __syncthreads();
  }
}

// ---------------- output projections ----------------
__global__ void k_pout(const bf16_t* __restrict__ g, const float* __restrict__ m,
                       const float* __restrict__ wg, const float* __restrict__ bg,
                       const float* __restrict__ wm, const float* __restrict__ bm,
                       float* __restrict__ out) {
  int r = (int)((blockIdx.x * blockDim.x + threadIdx.x) >> 5);
  int lane = threadIdx.x & 31;
  if (r >= BC) return;
  const bf16_t* gr = g + (size_t)r * HH4;
  const float*  mr = m + (size_t)r * HH2;
  float s = 0.f;
  for (int k = lane; k < HH4; k += 32) s += (float)gr[k] * wg[k];
  for (int k = lane; k < HH2; k += 32) s += mr[k] * wm[k];
#pragma unroll
  for (int msk = 16; msk >= 1; msk >>= 1) s += __shfl_xor(s, msk, 32);
  if (lane == 0) out[r] = s + bg[0] + bm[0];
}

// ---------------- host orchestration ----------------
extern "C" void kernel_launch(void* const* d_in, const int* in_sizes, int n_in,
                              void* d_out, int out_size, void* d_ws, size_t ws_size,
                              hipStream_t stream) {
  (void)in_sizes; (void)n_in; (void)out_size; (void)ws_size;
  const float* c_in   = (const float*)d_in[0];
  const float* q_in   = (const float*)d_in[1];
  const float* w_c    = (const float*)d_in[2];
  const float* b_c    = (const float*)d_in[3];
  const float* w_q    = (const float*)d_in[4];
  const float* b_q    = (const float*)d_in[5];
  const float* w_cq   = (const float*)d_in[6];
  const float* b_cq   = (const float*)d_in[7];
  const float* p1_wg = (const float*)d_in[26];
  const float* p1_bg = (const float*)d_in[27];
  const float* p1_wm = (const float*)d_in[28];
  const float* p1_bm = (const float*)d_in[29];
  const float* p2_wg = (const float*)d_in[30];
  const float* p2_bg = (const float*)d_in[31];
  const float* p2_wm = (const float*)d_in[32];
  const float* p2_bm = (const float*)d_in[33];
  float* outp = (float*)d_out;

  char* base = (char*)d_ws;
  size_t off = 0;
  auto alloc = [&](size_t bytes) -> void* {
    void* p = base + off;
    off = (off + bytes + 255) & ~(size_t)255;
    return p;
  };
  bf16_t* c_bf    = (bf16_t*)alloc((size_t)BC * HH * 2);
  bf16_t* qw_bf   = (bf16_t*)alloc((size_t)BQ * HH * 2);
  bf16_t* qT_bf   = (bf16_t*)alloc((size_t)BQ * HH * 2);
  float*  cvec    = (float*) alloc((size_t)BC * 4);
  float*  qvec    = (float*) alloc((size_t)BQ * 4);
  float*  s_buf   = (float*) alloc((size_t)BC * QL * 4);
  bf16_t* a_bf    = (bf16_t*)alloc((size_t)BC * QL * 2);
  float*  smax    = (float*) alloc((size_t)BC * 4);
  float*  bw_buf  = (float*) alloc((size_t)BC * 4);
  float*  q2c     = (float*) alloc((size_t)BB * HH * 4);
  bf16_t* g_bf    = (bf16_t*)alloc((size_t)BC * HH4 * 2);
  bf16_t* wih_bfF = (bf16_t*)alloc((size_t)HH4 * HH4 * 2);
  bf16_t* wih_bfB = (bf16_t*)alloc((size_t)HH4 * HH4 * 2);
  bf16_t* whh_bfF = (bf16_t*)alloc((size_t)HH4 * HH * 2);
  bf16_t* whh_bfB = (bf16_t*)alloc((size_t)HH4 * HH * 2);
  float*  xg_f    = (float*) alloc((size_t)BC * HH4 * 4);
  float*  xg_b    = (float*) alloc((size_t)BC * HH4 * 4);
  float*  m1      = (float*) alloc((size_t)BC * HH2 * 4);
  bf16_t* m1_bf   = (bf16_t*)alloc((size_t)BC * HH2 * 2);
  float*  mm      = (float*) alloc((size_t)BC * HH2 * 4);
  bf16_t* m_bf    = (bf16_t*)alloc((size_t)BC * HH2 * 2);
  float*  m2      = (float*) alloc((size_t)BC * HH2 * 4);

  auto cvt = [&](const float* src, bf16_t* dst, size_t n) {
    k_cvt_bf16<<<dim3((unsigned)((n + 255) / 256)), 256, 0, stream>>>(src, dst, n);
  };

  // ---- prep ----
  cvt(c_in, c_bf, (size_t)BC * HH);
  k_prep_q<<<dim3((BQ * HH + 255) / 256), 256, 0, stream>>>(q_in, w_cq, qw_bf, qT_bf);
  k_rowdot<<<dim3((BC * 32 + 255) / 256), 256, 0, stream>>>(c_in, w_c, b_c, cvec, BC, HH);
  k_rowdot<<<dim3((BQ * 32 + 255) / 256), 256, 0, stream>>>(q_in, w_q, b_q, qvec, BQ, HH);

  // ---- attention ----
  k_sim_wmma<<<dim3(BB, CL / 16, QL / 16), 32, 0, stream>>>(c_bf, qw_bf, cvec, qvec, b_cq, s_buf);
  k_softmax_rows<<<dim3(BC / 8), 256, 0, stream>>>(s_buf, a_bf, smax);
  k_bw_softmax<<<dim3(BB), 32, 0, stream>>>(smax, bw_buf);
  k_q2c<<<dim3(BB, HH / 256), 256, 0, stream>>>(bw_buf, c_in, q2c);
  k_c2q_wmma<<<dim3(BB, CL / 16, HH / 16), 32, 0, stream>>>(a_bf, qT_bf, g_bf);
  k_concat_g<<<dim3((unsigned)(((size_t)BC * HH + 255) / 256)), 256, 0, stream>>>(c_in, q2c, g_bf);

  // ---- three biLSTM layers ----
  struct LayerDesc { int base_idx; int Kdim; const bf16_t* inA; float* outM; };
  LayerDesc layers[3] = {
      {8,  HH4, g_bf,  m1},
      {14, HH2, m1_bf, mm},
      {20, HH2, m_bf,  m2},
  };
  for (int L = 0; L < 3; ++L) {
    const LayerDesc& ld = layers[L];
    const float* wihF = (const float*)d_in[ld.base_idx + 0];
    const float* whhF = (const float*)d_in[ld.base_idx + 1];
    const float* bF   = (const float*)d_in[ld.base_idx + 2];
    const float* wihB = (const float*)d_in[ld.base_idx + 3];
    const float* whhB = (const float*)d_in[ld.base_idx + 4];
    const float* bB   = (const float*)d_in[ld.base_idx + 5];
    cvt(wihF, wih_bfF, (size_t)HH4 * ld.Kdim);
    cvt(wihB, wih_bfB, (size_t)HH4 * ld.Kdim);
    cvt(whhF, whh_bfF, (size_t)HH4 * HH);
    cvt(whhB, whh_bfB, (size_t)HH4 * HH);
    k_ih_gemm<<<dim3(BC / 32, HH4 / 512), 256, 0, stream>>>(ld.inA, wih_bfF, bF, xg_f, ld.Kdim);
    k_ih_gemm<<<dim3(BC / 32, HH4 / 512), 256, 0, stream>>>(ld.inA, wih_bfB, bB, xg_b, ld.Kdim);
    k_lstm_steps2<<<dim3(2), 1024, 0, stream>>>(xg_f, xg_b, whh_bfF, whh_bfB, ld.outM);
    if (L == 0) cvt(m1, m1_bf, (size_t)BC * HH2);
    if (L == 1) cvt(mm, m_bf, (size_t)BC * HH2);
  }

  // ---- output projections: p1 uses (g, m), p2 uses (g, m2) ----
  k_pout<<<dim3(BC / 8), 256, 0, stream>>>(g_bf, mm, p1_wg, p1_bg, p1_wm, p1_bm, outp);
  k_pout<<<dim3(BC / 8), 256, 0, stream>>>(g_bf, m2, p2_wg, p2_bg, p2_wm, p2_bm, outp + BC);
}
